// VQVAE_29592324670280
// MI455X (gfx1250) — matrix operand order repaired and checked by reading, not compiled, not run on hardware
//
#include <hip/hip_runtime.h>
#include <hip/hip_bf16.h>
#include <math.h>

// ---------------------------------------------------------------------------
// CDNA5 WMMA types / helpers (wave32; ISA 7.12.2 fragment layouts)
// ---------------------------------------------------------------------------
typedef __attribute__((ext_vector_type(16))) __bf16        v16bf;
typedef __attribute__((ext_vector_type(8)))  float         v8f;
typedef __attribute__((ext_vector_type(4)))  unsigned int  v4u;

union Frag { v16bf v; unsigned short s[16]; v4u q[2]; };

__device__ __forceinline__ unsigned short f32_bf16(float f) {
    union { float f; unsigned int u; } c; c.f = f;
    unsigned int u = c.u;
    u += 0x7fffu + ((u >> 16) & 1u);          // round-to-nearest-even
    return (unsigned short)(u >> 16);
}
__device__ __forceinline__ float bf2f(unsigned short h) {
    union { unsigned int u; float f; } c; c.u = ((unsigned int)h) << 16;
    return c.f;
}

// ---------------------------------------------------------------------------
// Fast implicit-GEMM conv on WMMA bf16, N-blocked x4.
//   in  : bf16 NHWC  [B, Hin, Win, Cin]      (Cin % 32 == 0)
//   pw  : bf16 packed [Cout][kh*kw][Cin]     (tap-major, channel-minor)
//   out : bf16 NHWC  [Npix, Cout]            (Cout % 16 == 0, Npix % 64 == 0)
// One wave = 16(Cout) x 64(pixels): 4 accumulators sharing each A fragment.
// Inner loop per 32-wide K step: 2 A-loads + 8 B-loads (b128) -> 4 v_wmma.
// ---------------------------------------------------------------------------
template <int IN_RELU>
__global__ __launch_bounds__(32)
void conv_fast_wmma(const unsigned short* __restrict__ in,
                    const unsigned short* __restrict__ pw,
                    const float* __restrict__ bias,
                    const unsigned short* __restrict__ res,   // bf16 NHWC or null
                    unsigned short* __restrict__ out,
                    int Bn, int Cin, int Hin, int Win,
                    int Cout, int Hout, int Wout,
                    int kh, int kw, int s, int p, int out_relu)
{
    const int lane  = threadIdx.x;
    const int ncol  = lane & 15;
    const int HW    = Hout * Wout;
    const int Pbase = blockIdx.x * 64;
    int pb[4], py[4], px[4];
#pragma unroll
    for (int j = 0; j < 4; ++j) {
        int P = Pbase + j * 16 + ncol;
        pb[j] = P / HW;
        int rr = P % HW;
        py[j] = rr / Wout; px[j] = rr % Wout;
    }
    const int mg    = blockIdx.y * 16 + (lane & 15);
    const int aKoff = (lane >> 4) * 8;
    const int bKoff = (lane >> 4) * 8;
    const int Kd    = kh * kw * Cin;
    const unsigned short* arow = pw + mg * Kd;

    v8f acc[4] = {};
    int t = 0;
    for (int ky = 0; ky < kh; ++ky) {
        for (int kx = 0; kx < kw; ++kx, ++t) {
            bool valid[4];
            const unsigned short* brow[4];
#pragma unroll
            for (int j = 0; j < 4; ++j) {
                const int iy = py[j] * s - p + ky;
                const int ix = px[j] * s - p + kx;
                valid[j] = (iy >= 0) & (iy < Hin) & (ix >= 0) & (ix < Win);
                brow[j]  = in + (((pb[j] * Hin + iy) * Win + ix) * Cin);
            }
            const unsigned short* abase = arow + t * Cin;
            for (int ci = 0; ci < Cin; ci += 32) {
                Frag Af;
                Af.q[0] = *(const v4u*)(abase + ci + aKoff);
                Af.q[1] = *(const v4u*)(abase + ci + 16 + aKoff);
#pragma unroll
                for (int j = 0; j < 4; ++j) {
                    Frag Bf;
                    if (valid[j]) {
                        Bf.q[0] = *(const v4u*)(brow[j] + ci + bKoff);
                        Bf.q[1] = *(const v4u*)(brow[j] + ci + bKoff + 16);
                    } else {
                        Bf.q[0] = (v4u)0; Bf.q[1] = (v4u)0;
                    }
                    if (IN_RELU) {
#pragma unroll
                        for (int i = 0; i < 16; ++i)
                            if (Bf.s[i] & 0x8000u) Bf.s[i] = 0;
                    }
                    acc[j] = __builtin_amdgcn_wmma_f32_16x16x32_bf16(
                                 false, Af.v, false, Bf.v, (short)0, acc[j],
                                 false, false);
                }
            }
        }
    }

    // C/D layout: lane -> pixel column, VGPR v -> row m = 8*(lane>=16)+v.
    const int mb = blockIdx.y * 16 + (lane >> 4) * 8;
#pragma unroll
    for (int j = 0; j < 4; ++j) {
        const int P = Pbase + j * 16 + ncol;
        Frag Rf, Of;
        if (res) Rf.q[0] = *(const v4u*)(res + P * Cout + mb);
#pragma unroll
        for (int v = 0; v < 8; ++v) {
            float val = acc[j][v] + bias[mb + v];
            if (res) val += bf2f(Rf.s[v]);
            if (out_relu) val = val > 0.0f ? val : 0.0f;
            Of.s[v] = f32_bf16(val);
        }
        *(v4u*)(out + P * Cout + mb) = Of.q[0];
    }
}

// ---------------------------------------------------------------------------
// ConvTranspose2d(k4,s2,p1) as 4 parity-class implicit GEMMs (K = Cin*4),
// N-blocked x4.  ppw : bf16 [4(parity)][Coutp][4(tap)][Cin], Coutp % 16 == 0.
// Output either bf16 NHWC (stride Coutp) or f32 NCHW (Cr real channels).
// act: 0 none, 1 relu, 2 sigmoid
// ---------------------------------------------------------------------------
template <int IN_RELU>
__global__ __launch_bounds__(32)
void deconv_fast_wmma(const unsigned short* __restrict__ in,
                      const unsigned short* __restrict__ ppw,
                      const float* __restrict__ bias,
                      unsigned short* __restrict__ out_bf,
                      float* __restrict__ out_f32,
                      int Bn, int Cin, int Hin, int Win,
                      int Coutp, int Cr, int act, int f32_out)
{
    const int lane  = threadIdx.x;
    const int ncol  = lane & 15;
    const int par   = blockIdx.z;
    const int py    = par >> 1, px = par & 1;
    const int HW    = Hin * Win;
    const int Pbase = blockIdx.x * 64;
    int pb[4], yy[4], xx[4];
#pragma unroll
    for (int j = 0; j < 4; ++j) {
        int P = Pbase + j * 16 + ncol;
        pb[j] = P / HW;
        int rr = P % HW;
        yy[j] = rr / Win; xx[j] = rr % Win;
    }
    const int mg    = blockIdx.y * 16 + (lane & 15);
    const int aKoff = (lane >> 4) * 8;
    const int bKoff = (lane >> 4) * 8;
    const unsigned short* arow = ppw + ((par * Coutp + mg) * 4) * Cin;

    v8f acc[4] = {};
#pragma unroll
    for (int t = 0; t < 4; ++t) {
        const int ty = t >> 1, tx = t & 1;
        bool valid[4];
        const unsigned short* brow[4];
#pragma unroll
        for (int j = 0; j < 4; ++j) {
            const int iy = yy[j] + py - ty;
            const int ix = xx[j] + px - tx;
            valid[j] = (iy >= 0) & (iy < Hin) & (ix >= 0) & (ix < Win);
            brow[j]  = in + (((pb[j] * Hin + iy) * Win + ix) * Cin);
        }
        const unsigned short* abase = arow + t * Cin;
        for (int ci = 0; ci < Cin; ci += 32) {
            Frag Af;
            Af.q[0] = *(const v4u*)(abase + ci + aKoff);
            Af.q[1] = *(const v4u*)(abase + ci + 16 + aKoff);
#pragma unroll
            for (int j = 0; j < 4; ++j) {
                Frag Bf;
                if (valid[j]) {
                    Bf.q[0] = *(const v4u*)(brow[j] + ci + bKoff);
                    Bf.q[1] = *(const v4u*)(brow[j] + ci + bKoff + 16);
                } else {
                    Bf.q[0] = (v4u)0; Bf.q[1] = (v4u)0;
                }
                if (IN_RELU) {
#pragma unroll
                    for (int i = 0; i < 16; ++i)
                        if (Bf.s[i] & 0x8000u) Bf.s[i] = 0;
                }
                acc[j] = __builtin_amdgcn_wmma_f32_16x16x32_bf16(
                             false, Af.v, false, Bf.v, (short)0, acc[j],
                             false, false);
            }
        }
    }

    const int mb   = blockIdx.y * 16 + (lane >> 4) * 8;
    const int Hout = 2 * Hin, Wout = 2 * Win;
#pragma unroll
    for (int j = 0; j < 4; ++j) {
        const int oy = 2 * yy[j] + py, ox = 2 * xx[j] + px;
        if (f32_out) {
#pragma unroll
            for (int v = 0; v < 8; ++v) {
                int m = mb + v;
                if (m < Cr) {
                    float val = acc[j][v] + bias[m];
                    if (act == 1)      val = val > 0.0f ? val : 0.0f;
                    else if (act == 2) val = 1.0f / (1.0f + expf(-val));
                    out_f32[((pb[j] * Cr + m) * Hout + oy) * Wout + ox] = val;
                }
            }
        } else {
            Frag Of;
#pragma unroll
            for (int v = 0; v < 8; ++v) {
                float val = acc[j][v] + bias[mb + v];
                if (act == 1) val = val > 0.0f ? val : 0.0f;
                Of.s[v] = f32_bf16(val);
            }
            *(v4u*)(out_bf + ((pb[j] * Hout + oy) * Wout + ox) * Coutp + mb) = Of.q[0];
        }
    }
}

// ---------------------------------------------------------------------------
// VQ argmin via WMMA, row-blocked x2: dist = ||cb||^2 - 2 * z.cb.
// z : bf16 [N,64], cbb : bf16 [512,64].  One wave = 32 z-rows x all 512 codes;
// each codebook tile load feeds 4 wmma.
// ---------------------------------------------------------------------------
__global__ __launch_bounds__(32)
void vq_argmin_wmma(const unsigned short* __restrict__ z,
                    const unsigned short* __restrict__ cbb,
                    const float* __restrict__ cbn, int* __restrict__ idx, int N)
{
    const int lane  = threadIdx.x;
    const int ncol  = lane & 15;
    const int aKoff = (lane >> 4) * 8;
    const int bKoff = (lane >> 4) * 8;

    Frag A0[2], A1[2];
#pragma unroll
    for (int g = 0; g < 2; ++g) {
        const unsigned short* zr = z + (blockIdx.x * 32 + g * 16 + (lane & 15)) * 64;
        A0[g].q[0] = *(const v4u*)(zr + aKoff);
        A0[g].q[1] = *(const v4u*)(zr + 16 + aKoff);
        A1[g].q[0] = *(const v4u*)(zr + 32 + aKoff);
        A1[g].q[1] = *(const v4u*)(zr + 48 + aKoff);
    }

    float best[2][8]; int bidx[2][8];
#pragma unroll
    for (int g = 0; g < 2; ++g)
#pragma unroll
        for (int v = 0; v < 8; ++v) { best[g][v] = 3.4e38f; bidx[g][v] = 0; }

    for (int ct = 0; ct < 32; ++ct) {
        const int cg = ct * 16 + ncol;
        const unsigned short* cr = cbb + cg * 64;
        Frag B0, B1;
        B0.q[0] = *(const v4u*)(cr + bKoff);
        B0.q[1] = *(const v4u*)(cr + bKoff + 16);
        B1.q[0] = *(const v4u*)(cr + 32 + bKoff);
        B1.q[1] = *(const v4u*)(cr + 32 + bKoff + 16);
        const float cn = cbn[cg];
#pragma unroll
        for (int g = 0; g < 2; ++g) {
            v8f acc = {};
            acc = __builtin_amdgcn_wmma_f32_16x16x32_bf16(
                      false, A0[g].v, false, B0.v, (short)0, acc, false, false);
            acc = __builtin_amdgcn_wmma_f32_16x16x32_bf16(
                      false, A1[g].v, false, B1.v, (short)0, acc, false, false);
#pragma unroll
            for (int v = 0; v < 8; ++v) {
                float d = cn - 2.0f * acc[v];
                if (d < best[g][v]) { best[g][v] = d; bidx[g][v] = cg; }
            }
        }
    }

    // Reduce across the 16 lanes of each half-wave (same 8-row set per half).
#pragma unroll
    for (int off = 8; off >= 1; off >>= 1) {
#pragma unroll
        for (int g = 0; g < 2; ++g)
#pragma unroll
            for (int v = 0; v < 8; ++v) {
                float ob = __shfl_xor(best[g][v], off, 16);
                int   oi = __shfl_xor(bidx[g][v], off, 16);
                if (ob < best[g][v] || (ob == best[g][v] && oi < bidx[g][v])) {
                    best[g][v] = ob; bidx[g][v] = oi;
                }
            }
    }
    if ((lane & 15) == 0) {
#pragma unroll
        for (int g = 0; g < 2; ++g) {
            int rbase = blockIdx.x * 32 + g * 16 + (lane >> 4) * 8;
#pragma unroll
            for (int v = 0; v < 8; ++v) idx[rbase + v] = bidx[g][v];
        }
    }
}

// ---------------------------------------------------------------------------
// enc_c1 im2col: f32 NCHW x [32,3,256,256] -> bf16 [524288 pixels][64],
// K-order k = tap*3 + ci (tap = ky*4+kx, k4 s2 p1), zero-padded to 64.
// One-off bandwidth-bound pass; lets enc_c1 use the standard fast GEMM.
// ---------------------------------------------------------------------------
__global__ void im2col_ec1_k(const float* __restrict__ x,
                             unsigned short* __restrict__ o, int Bn)
{
    int i = blockIdx.x * blockDim.x + threadIdx.x;   // over Npix*64
    int total = Bn * 128 * 128 * 64;
    if (i >= total) return;
    int n = i >> 6, k = i & 63;
    float v = 0.0f;
    if (k < 48) {
        int t  = k / 3, ci = k - t * 3;
        int ky = t >> 2, kx = t & 3;
        int b  = n >> 14;                 // / (128*128)
        int rr = n & 16383;
        int oy = rr >> 7, ox = rr & 127;
        int iy = oy * 2 - 1 + ky, ix = ox * 2 - 1 + kx;
        if (iy >= 0 && iy < 256 && ix >= 0 && ix < 256)
            v = x[((b * 3 + ci) * 256 + iy) * 256 + ix];
    }
    o[i] = f32_bf16(v);
}

__global__ void pack_ec1_k(const float* __restrict__ w, unsigned short* __restrict__ o)
{
    int i = blockIdx.x * blockDim.x + threadIdx.x;   // over 64*64
    if (i >= 64 * 64) return;
    int m = i >> 6, k = i & 63;
    float v = 0.0f;
    if (k < 48) {
        int t  = k / 3, ci = k - t * 3;
        int ky = t >> 2, kx = t & 3;
        v = w[((m * 3 + ci) * 4 + ky) * 4 + kx];
    }
    o[i] = f32_bf16(v);
}

// ---------------------------------------------------------------------------
// Packing / support kernels
// ---------------------------------------------------------------------------
__global__ void pack_w_k(const float* __restrict__ w, unsigned short* __restrict__ o,
                         int Cout, int Cin, int kh, int kw) {
    int i = blockIdx.x * blockDim.x + threadIdx.x;
    int KK = kh * kw;
    int total = Cout * KK * Cin;
    if (i >= total) return;
    int m = i / (KK * Cin);
    int r = i % (KK * Cin);
    int t = r / Cin, ci = r % Cin;
    int ky = t / kw, kx = t % kw;
    o[i] = f32_bf16(w[((m * Cin + ci) * kh + ky) * kw + kx]);
}

__global__ void pack_dw_k(const float* __restrict__ w, unsigned short* __restrict__ o,
                          int Cr, int Coutp, int Cin) {
    int i = blockIdx.x * blockDim.x + threadIdx.x;
    int total = 4 * Coutp * 4 * Cin;
    if (i >= total) return;
    int par = i / (Coutp * 4 * Cin);
    int r   = i % (Coutp * 4 * Cin);
    int m   = r / (4 * Cin);
    int r2  = r % (4 * Cin);
    int t   = r2 / Cin, ci = r2 % Cin;
    int py = par >> 1, px = par & 1, ty = t >> 1, tx = t & 1;
    int ky = 1 - py + 2 * ty, kx = 1 - px + 2 * tx;
    float v = (m < Cr) ? w[((ci * Cr + m) * 4 + ky) * 4 + kx] : 0.0f;
    o[i] = f32_bf16(v);
}

__global__ void pack_cb_k(const float* __restrict__ cb, unsigned short* __restrict__ o) {
    int i = blockIdx.x * blockDim.x + threadIdx.x;
    if (i < 512 * 64) o[i] = f32_bf16(cb[i]);
}

__global__ void zero_k(float* __restrict__ p, int n) {
    int i = blockIdx.x * blockDim.x + threadIdx.x;
    if (i < n) p[i] = 0.0f;
}

__global__ void cbnorm_k(const float* __restrict__ cb, float* __restrict__ cbn) {
    int k = blockIdx.x * blockDim.x + threadIdx.x;
    if (k < 512) {
        float s = 0.0f;
        for (int d = 0; d < 64; ++d) { float c = cb[k * 64 + d]; s += c * c; }
        cbn[k] = s;
    }
}

// Gather q = cb[idx] (bf16 [N,64] for the decoder), accumulate SSE + histogram.
__global__ void vq_gather_k(const unsigned short* __restrict__ z,
                            const float* __restrict__ cb,
                            const int* __restrict__ idx,
                            unsigned short* __restrict__ q,
                            float* __restrict__ sse, float* __restrict__ counts,
                            int N)
{
    int n = blockIdx.x * blockDim.x + threadIdx.x;
    if (n >= N) return;
    int id = idx[n];
    float local = 0.0f;
    for (int d = 0; d < 64; ++d) {
        float qv = cb[id * 64 + d];
        float zv = bf2f(z[n * 64 + d]);
        float df = qv - zv;
        local += df * df;
        q[n * 64 + d] = f32_bf16(qv);
    }
    atomicAdd(sse, local);
    atomicAdd(&counts[id], 1.0f);
}

// loss = (1+beta)*mse ; perplexity = exp(-sum p log p)
__global__ void finalize_k(const float* __restrict__ sse,
                           const float* __restrict__ counts,
                           float* __restrict__ out_loss,
                           float* __restrict__ out_ppl, int N)
{
    __shared__ float sh[512];
    int t = threadIdx.x;
    float p = counts[t] / (float)N;
    sh[t] = -p * logf(p + 1e-10f);
    __syncthreads();
    for (int s = 256; s > 0; s >>= 1) {
        if (t < s) sh[t] += sh[t + s];
        __syncthreads();
    }
    if (t == 0) {
        *out_ppl  = expf(sh[0]);
        *out_loss = 1.25f * (*sse) / ((float)N * 64.0f);
    }
}

// ---------------------------------------------------------------------------
// Orchestration
// ---------------------------------------------------------------------------
extern "C" void kernel_launch(void* const* d_in, const int* in_sizes, int n_in,
                              void* d_out, int out_size, void* d_ws, size_t ws_size,
                              hipStream_t stream)
{
    (void)in_sizes; (void)n_in; (void)ws_size;
    const float* x     = (const float*)d_in[0];
    const float* ec1w  = (const float*)d_in[1];   const float* ec1b  = (const float*)d_in[2];
    const float* ec2w  = (const float*)d_in[3];   const float* ec2b  = (const float*)d_in[4];
    const float* ec3w  = (const float*)d_in[5];   const float* ec3b  = (const float*)d_in[6];
    const float* er0w1 = (const float*)d_in[7];   const float* er0b1 = (const float*)d_in[8];
    const float* er0w2 = (const float*)d_in[9];   const float* er0b2 = (const float*)d_in[10];
    const float* er1w1 = (const float*)d_in[11];  const float* er1b1 = (const float*)d_in[12];
    const float* er1w2 = (const float*)d_in[13];  const float* er1b2 = (const float*)d_in[14];
    const float* pvqw  = (const float*)d_in[15];  const float* pvqb  = (const float*)d_in[16];
    const float* cb    = (const float*)d_in[17];
    const float* dc1w  = (const float*)d_in[18];  const float* dc1b  = (const float*)d_in[19];
    const float* dr0w1 = (const float*)d_in[20];  const float* dr0b1 = (const float*)d_in[21];
    const float* dr0w2 = (const float*)d_in[22];  const float* dr0b2 = (const float*)d_in[23];
    const float* dr1w1 = (const float*)d_in[24];  const float* dr1b1 = (const float*)d_in[25];
    const float* dr1w2 = (const float*)d_in[26];  const float* dr1b2 = (const float*)d_in[27];
    const float* dd1w  = (const float*)d_in[28];  const float* dd1b  = (const float*)d_in[29];
    const float* dd2w  = (const float*)d_in[30];  const float* dd2b  = (const float*)d_in[31];

    float* dout = (float*)d_out;

    // ---- workspace layout (bytes) ----
    char* wsb = (char*)d_ws;
    unsigned short* usA  = (unsigned short*)wsb;               // 33,554,432 bf16 (enc_c1 / dd1 out)
    unsigned short* usR  = usA;                                //  4,194,304 bf16 overlay (res hidden)
    unsigned short* usIM = (unsigned short*)(wsb + 67108864);  // 33,554,432 bf16 (enc_c1 im2col; dies before enc_c2)
    unsigned short* usB  = (unsigned short*)(wsb + 67108864);  // 16,777,216 bf16 (enc_c2 out)
    unsigned short* usZ  = usB;                                //  8,388,608 bf16 (z, [N,64])
    unsigned short* usQ  = usB + 8388608;                      //  8,388,608 bf16 (q, [N,64])
    unsigned short* usC  = (unsigned short*)(wsb + 100663296); // 16,777,216 bf16 (enc_c3 / res / dec_c1)
    float* fCBN = (float*)(wsb + 134217728);                   // 512
    float* fCNT = fCBN + 512;                                  // 512
    float* fSSE = fCNT + 512;                                  // 1
    int*   iIDX = (int*)(fSSE + 1);                            // 131,072
    unsigned short* pk = (unsigned short*)(wsb + 134746128);   // packed weights (16B aligned)
    unsigned short* pw_ec1  = pk; pk += 4096;
    unsigned short* pw_ec2  = pk; pk += 131072;
    unsigned short* pw_ec3  = pk; pk += 147456;
    unsigned short* pw_er0a = pk; pk += 36864;
    unsigned short* pw_er0b = pk; pk += 4096;
    unsigned short* pw_er1a = pk; pk += 36864;
    unsigned short* pw_er1b = pk; pk += 4096;
    unsigned short* pw_pvq  = pk; pk += 8192;
    unsigned short* pw_dc1  = pk; pk += 73728;
    unsigned short* pw_dr0a = pk; pk += 36864;
    unsigned short* pw_dr0b = pk; pk += 4096;
    unsigned short* pw_dr1a = pk; pk += 36864;
    unsigned short* pw_dr1b = pk; pk += 4096;
    unsigned short* pp_dd1  = pk; pk += 131072;
    unsigned short* pp_dd2  = pk; pk += 16384;
    unsigned short* cbbf    = pk; pk += 32768;

    const int Bn = 32;
    const int N  = Bn * 64 * 64;    // 131072 latent positions

    // ---- pack weights / codebook to bf16 GEMM layouts ----
    auto packw = [&](const float* w, unsigned short* o, int Co, int Ci, int kh, int kw) {
        int tot = Co * kh * kw * Ci;
        pack_w_k<<<(tot + 255) / 256, 256, 0, stream>>>(w, o, Co, Ci, kh, kw);
    };
    pack_ec1_k<<<(64 * 64 + 255) / 256, 256, 0, stream>>>(ec1w, pw_ec1);
    packw(ec2w,  pw_ec2,  128, 64, 4, 4);
    packw(ec3w,  pw_ec3,  128, 128, 3, 3);
    packw(er0w1, pw_er0a, 32, 128, 3, 3);   packw(er0w2, pw_er0b, 128, 32, 1, 1);
    packw(er1w1, pw_er1a, 32, 128, 3, 3);   packw(er1w2, pw_er1b, 128, 32, 1, 1);
    packw(pvqw,  pw_pvq,  64, 128, 1, 1);
    packw(dc1w,  pw_dc1,  128, 64, 3, 3);
    packw(dr0w1, pw_dr0a, 32, 128, 3, 3);   packw(dr0w2, pw_dr0b, 128, 32, 1, 1);
    packw(dr1w1, pw_dr1a, 32, 128, 3, 3);   packw(dr1w2, pw_dr1b, 128, 32, 1, 1);
    pack_dw_k<<<(4 * 64 * 4 * 128 + 255) / 256, 256, 0, stream>>>(dd1w, pp_dd1, 64, 64, 128);
    pack_dw_k<<<(4 * 16 * 4 * 64  + 255) / 256, 256, 0, stream>>>(dd2w, pp_dd2, 3, 16, 64);
    pack_cb_k<<<(512 * 64 + 255) / 256, 256, 0, stream>>>(cb, cbbf);
    cbnorm_k<<<2, 256, 0, stream>>>(cb, fCBN);
    zero_k<<<3, 256, 0, stream>>>(fCNT, 513);

    auto conv = [&](const unsigned short* in, const unsigned short* pwg,
                    const float* b, const unsigned short* res, unsigned short* out,
                    int Ci, int Hi, int Wi, int Co, int Ho, int Wo,
                    int kh, int kw, int s, int p, int ir, int orl) {
        int Npix = Bn * Ho * Wo;
        dim3 grid(Npix / 64, Co / 16, 1);
        if (ir)
            conv_fast_wmma<1><<<grid, 32, 0, stream>>>(in, pwg, b, res, out,
                Bn, Ci, Hi, Wi, Co, Ho, Wo, kh, kw, s, p, orl);
        else
            conv_fast_wmma<0><<<grid, 32, 0, stream>>>(in, pwg, b, res, out,
                Bn, Ci, Hi, Wi, Co, Ho, Wo, kh, kw, s, p, orl);
    };

    // ---- encoder ----
    // enc_c1: im2col (bf16, K padded 48->64) then standard fast GEMM (k=1).
    im2col_ec1_k<<<(Bn * 128 * 128 * 64 + 255) / 256, 256, 0, stream>>>(x, usIM, Bn);
    conv(usIM, pw_ec1, ec1b, nullptr, usA, 64, 128, 128, 64, 128, 128, 1, 1, 1, 0, 0, 1);
    conv(usA, pw_ec2, ec2b, nullptr, usB, 64, 128, 128, 128, 64, 64, 4, 4, 2, 1, 0, 1);
    conv(usB, pw_ec3, ec3b, nullptr, usC, 128, 64, 64, 128, 64, 64, 3, 3, 1, 1, 0, 1);
    // enc residual stack (in-place on usC)
    conv(usC, pw_er0a, er0b1, nullptr, usR, 128, 64, 64, 32,  64, 64, 3, 3, 1, 1, 1, 1);
    conv(usR, pw_er0b, er0b2, usC,     usC, 32,  64, 64, 128, 64, 64, 1, 1, 1, 0, 0, 0);
    conv(usC, pw_er1a, er1b1, nullptr, usR, 128, 64, 64, 32,  64, 64, 3, 3, 1, 1, 1, 1);
    conv(usR, pw_er1b, er1b2, usC,     usC, 32,  64, 64, 128, 64, 64, 1, 1, 1, 0, 0, 0);
    // pre_vq (stack-final relu folded into input)
    conv(usC, pw_pvq, pvqb, nullptr, usZ, 128, 64, 64, 64, 64, 64, 1, 1, 1, 0, 1, 0);

    // ---- vector quantization ----
    vq_argmin_wmma<<<N / 32, 32, 0, stream>>>(usZ, cbbf, fCBN, iIDX, N);
    vq_gather_k<<<(N + 255) / 256, 256, 0, stream>>>(usZ, cb, iIDX, usQ, fSSE, fCNT, N);

    // ---- decoder ----
    conv(usQ, pw_dc1, dc1b, nullptr, usC, 64, 64, 64, 128, 64, 64, 3, 3, 1, 1, 0, 0);
    conv(usC, pw_dr0a, dr0b1, nullptr, usR, 128, 64, 64, 32,  64, 64, 3, 3, 1, 1, 1, 1);
    conv(usR, pw_dr0b, dr0b2, usC,     usC, 32,  64, 64, 128, 64, 64, 1, 1, 1, 0, 0, 0);
    conv(usC, pw_dr1a, dr1b1, nullptr, usR, 128, 64, 64, 32,  64, 64, 3, 3, 1, 1, 1, 1);
    conv(usR, pw_dr1b, dr1b2, usC,     usC, 32,  64, 64, 128, 64, 64, 1, 1, 1, 0, 0, 0);

    // dec_d1: deconv 128->64, 64x64 -> 128x128, relu, bf16 NHWC out (in_relu folded)
    {
        dim3 grid((Bn * 64 * 64) / 64, 64 / 16, 4);
        deconv_fast_wmma<1><<<grid, 32, 0, stream>>>(usC, pp_dd1, dd1b, usA, nullptr,
                                                     Bn, 128, 64, 64, 64, 64, 1, 0);
    }
    // dec_d2: deconv 64->3 (padded 16), 128x128 -> 256x256, sigmoid, f32 NCHW -> d_out[1..]
    {
        dim3 grid((Bn * 128 * 128) / 64, 16 / 16, 4);
        deconv_fast_wmma<0><<<grid, 32, 0, stream>>>(usA, pp_dd2, dd2b, nullptr, dout + 1,
                                                     Bn, 64, 128, 128, 16, 3, 2, 1);
    }

    // ---- scalars: loss -> d_out[0], perplexity -> d_out[out_size-1] ----
    finalize_k<<<1, 512, 0, stream>>>(fSSE, fCNT, dout, dout + (out_size - 1), N);
}
